// LSTMCodingOptimizer_3762391351481
// MI455X (gfx1250) — compile-verified, matrix operand order checked by reading, not run on hardware
//
#include <hip/hip_runtime.h>
#include <hip/hip_bf16.h>
#include <math.h>

constexpr int cV = 32000;
constexpr int cH = 256;
constexpr int cB = 16;
constexpr int cS = 256;
constexpr int cT = 64;
constexpr int cG = 4 * cH;   // 1024 gate width

typedef __attribute__((ext_vector_type(16))) _Float16 v16h;
typedef __attribute__((ext_vector_type(8)))  _Float16 v8h;
typedef __attribute__((ext_vector_type(8)))  float    v8f;

union HV16 { v16h v; v8h h[2]; };

__device__ __forceinline__ float sigf(float x) { return 1.0f / (1.0f + __expf(-x)); }

__device__ __forceinline__ v8f wmma_f16(const HV16& a, const HV16& b, v8f c)
{
  return __builtin_amdgcn_wmma_f32_16x16x32_f16(false, a.v, false, b.v,
                                                (short)0, c, false, false);
}

// ---------------- generic batched WMMA GEMM:  C = A @ W^T (+bias) ----------------
// A: (M,K) f16 row-major.  W: (N,K) f16 row-major.  C: (M,N) f32 or f16.
// One wave owns a 16x64 C strip (4 N-tiles sharing one A fragment) so each
// A load feeds 4 v_wmma_f32_16x16x32_f16 ops; K advances 32/step.
template<int ACT, int HALF_OUT>
__global__ __launch_bounds__(256) void gemm_wmma(
    const _Float16* __restrict__ A, long sA,
    const _Float16* __restrict__ W, long sW,
    const float* __restrict__ bias,
    void* __restrict__ Cout, long sC,
    int M, int N, int K)
{
  const int batch = blockIdx.y;
  const _Float16* Ab = A + (long)batch * sA;
  const _Float16* Wb = W + (long)batch * sW;
  int grp  = blockIdx.x * (blockDim.x >> 5) + (threadIdx.x >> 5);
  int grpN = N >> 6;                       // groups of 4 N-tiles
  int tm = grp / grpN;
  if (tm >= (M >> 4)) return;              // wave-uniform guard
  int tn0 = (grp - tm * grpN) << 2;        // first of 4 N-tiles
  int lane = threadIdx.x & 31;
  int r  = lane & 15;                      // row (A) / col (B) within tile
  int hi = lane >> 4;                      // half-wave select
  int lo = hi << 3;                        // K offset 0 or 8
  const _Float16* arow = Ab + (long)(tm * 16 + r) * K + lo;
  const _Float16* brow = Wb + (long)(tn0 * 16 + r) * K + lo;
  const long bstr = (long)16 * K;
  v8f acc[4] = {};
  for (int k = 0; k < K; k += 32) {
    HV16 a, b0, b1, b2, b3;
    a.h[0]  = *(const v8h*)(arow + k);
    a.h[1]  = *(const v8h*)(arow + k + 16);
    b0.h[0] = *(const v8h*)(brow + k);
    b0.h[1] = *(const v8h*)(brow + k + 16);
    b1.h[0] = *(const v8h*)(brow + bstr + k);
    b1.h[1] = *(const v8h*)(brow + bstr + k + 16);
    b2.h[0] = *(const v8h*)(brow + 2 * bstr + k);
    b2.h[1] = *(const v8h*)(brow + 2 * bstr + k + 16);
    b3.h[0] = *(const v8h*)(brow + 3 * bstr + k);
    b3.h[1] = *(const v8h*)(brow + 3 * bstr + k + 16);
    acc[0] = wmma_f16(a, b0, acc[0]);
    acc[1] = wmma_f16(a, b1, acc[1]);
    acc[2] = wmma_f16(a, b2, acc[2]);
    acc[3] = wmma_f16(a, b3, acc[3]);
  }
  int m0 = tm * 16 + hi * 8;
  #pragma unroll
  for (int t = 0; t < 4; ++t) {
    int n = (tn0 + t) * 16 + r;
    float bv = bias ? bias[n] : 0.0f;
    #pragma unroll
    for (int i = 0; i < 8; ++i) {
      float vv = acc[t][i] + bv;
      if (ACT) vv = fmaxf(vv, 0.0f);
      if (HALF_OUT)
        ((_Float16*)Cout)[(long)batch * sC + (long)(m0 + i) * N + n] = (_Float16)vv;
      else
        ((float*)Cout)[(long)batch * sC + (long)(m0 + i) * N + n] = vv;
    }
  }
}

// ---------------- persistent single-WGP LSTM recurrence -------------------------
// Gx: (L,B,4H) f32 precomputed x@Wih^T + b.  Whh: (4H,H) f16.
// h (f16) lives in LDS (row-padded to break bank stride), c in registers.
// Per step: cache the wave-shared A fragments (h) in 64 VGPRs once, then walk
// 8 N-tiles with only one B fragment live (no spills), 8 WMMAs each.
__global__ __launch_bounds__(256) void lstm_wmma(
    const float* __restrict__ Gx,
    const _Float16* __restrict__ Whh,
    const float* __restrict__ h0,
    const float* __restrict__ c0,
    float* __restrict__ hs,           // (L,B,H) f32
    float* __restrict__ hfin,         // (B,H) or null
    float* __restrict__ cfin,
    int L, int rev)
{
  constexpr int HP = cH + 8;
  __shared__ alignas(16) _Float16 hsh[cB * HP];   // 8.25 KB
  __shared__ _Float16 gsh[cB * cG];               // 32 KB
  const int tid = threadIdx.x;
  float creg[16];
  #pragma unroll
  for (int j = 0; j < 16; ++j) {
    int i = tid + 256 * j;
    int b = i >> 8, h = i & 255;
    hsh[b * HP + h] = (_Float16)h0[i];
    creg[j] = c0[i];
  }
  __syncthreads();
  const int wave = tid >> 5, lane = tid & 31;
  const int r = lane & 15, hi = lane >> 4, lo = hi << 3;
  for (int l = 0; l < L; ++l) {
    // cache A fragments for the whole step (same for all 8 tiles of this wave)
    HV16 af[8];
    #pragma unroll
    for (int kk = 0; kk < 8; ++kk) {
      const _Float16* arow = &hsh[r * HP + lo + kk * 32];
      af[kk].h[0] = *(const v8h*)(arow);
      af[kk].h[1] = *(const v8h*)(arow + 16);
    }
    #pragma unroll 1
    for (int t8 = 0; t8 < 8; ++t8) {
      int tn = wave * 8 + t8;
      const _Float16* brow = Whh + (tn * 16 + r) * cH + lo;
      v8f acc = {};
      #pragma unroll
      for (int kk = 0; kk < 8; ++kk) {
        HV16 b;
        b.h[0] = *(const v8h*)(brow + kk * 32);
        b.h[1] = *(const v8h*)(brow + kk * 32 + 16);
        acc = wmma_f16(af[kk], b, acc);
      }
      int n = tn * 16 + r, m0 = hi * 8;
      #pragma unroll
      for (int i = 0; i < 8; ++i)
        gsh[(m0 + i) * cG + n] = (_Float16)acc[i];
    }
    __syncthreads();
    const int gl = rev ? (L - 1 - l) : l;
    const float* gx = Gx + (long)gl * cB * cG;
    #pragma unroll
    for (int j = 0; j < 16; ++j) {
      int i = tid + 256 * j;
      int b = i >> 8, h = i & 255;
      const float*    gxr = gx  + b * cG;
      const _Float16* gsr = gsh + b * cG;
      float gi = (float)gsr[h]           + gxr[h];
      float gf = (float)gsr[cH + h]      + gxr[cH + h];
      float gg = (float)gsr[2 * cH + h]  + gxr[2 * cH + h];
      float go = (float)gsr[3 * cH + h]  + gxr[3 * cH + h];
      float c  = sigf(gf) * creg[j] + sigf(gi) * tanhf(gg);
      float hh = sigf(go) * tanhf(c);
      creg[j] = c;
      hs[(long)l * cB * cH + i] = hh;
      hsh[b * HP + h] = (_Float16)hh;
    }
    __syncthreads();
  }
  if (hfin) {
    #pragma unroll
    for (int j = 0; j < 16; ++j) {
      int i = tid + 256 * j;
      hfin[i] = (float)hsh[(i >> 8) * HP + (i & 255)];
      cfin[i] = creg[j];
    }
  }
}

// ---------------- small elementwise / reduction kernels -------------------------
__global__ __launch_bounds__(256) void embed_gather(
    const int* __restrict__ ids, const float* __restrict__ emb,
    _Float16* __restrict__ out, int Len)   // out (Len,B,H)
{
  int idx = blockIdx.x * blockDim.x + threadIdx.x;
  if (idx >= Len * cB * cH) return;
  int h = idx & 255, rb = idx >> 8;
  int b = rb & 15, s = rb >> 4;
  int tok = ids[b * Len + s];
  out[idx] = (_Float16)emb[(long)tok * cH + h];
}

__global__ __launch_bounds__(256) void cvt_f16(
    const float* __restrict__ src, _Float16* __restrict__ dst,
    int rows, int cols, int stride)
{
  int idx = blockIdx.x * blockDim.x + threadIdx.x;
  if (idx >= rows * cols) return;
  int r = idx / cols, c = idx - r * cols;
  dst[idx] = (_Float16)src[(long)r * stride + c];
}

__global__ __launch_bounds__(256) void fill_zero(float* p, int n)
{ int i = blockIdx.x * blockDim.x + threadIdx.x; if (i < n) p[i] = 0.0f; }

__global__ __launch_bounds__(256) void enc_combine(
    const float* __restrict__ hf, const float* __restrict__ hb,
    float* __restrict__ enc32, _Float16* __restrict__ enc16)
{
  int idx = blockIdx.x * blockDim.x + threadIdx.x;   // (b,s,h)
  if (idx >= cB * cS * cH) return;
  int h = idx & 255, bs = idx >> 8;
  int s = bs & 255, b = bs >> 8;
  float v = hf[(s * cB + b) * cH + h] + hb[((cS - 1 - s) * cB + b) * cH + h];
  enc32[idx] = v;
  enc16[idx] = (_Float16)v;
}

__global__ __launch_bounds__(256) void tbh_transpose(
    const float* __restrict__ hsTB,  // (T,B,H) -> (B,T,H)
    float* __restrict__ o32, _Float16* __restrict__ o16)
{
  int idx = blockIdx.x * blockDim.x + threadIdx.x;
  if (idx >= cB * cT * cH) return;
  int h = idx & 255, bt = idx >> 8;
  int t = bt & 63, b = bt >> 6;
  float v = hsTB[(t * cB + b) * cH + h];
  o32[idx] = v; o16[idx] = (_Float16)v;
}

__global__ __launch_bounds__(256) void mode_kernel(
    const float* __restrict__ dh, const float* __restrict__ mW,
    const float* __restrict__ mb, float* __restrict__ mode0)
{
  int i = blockIdx.x * blockDim.x + threadIdx.x;
  if (i >= cB * cT) return;
  const float* row = dh + (long)i * cH;
  float a0 = mb[0], a1 = mb[1];
  for (int h = 0; h < cH; ++h) { float x = row[h]; a0 += x * mW[h]; a1 += x * mW[cH + h]; }
  float m = fmaxf(a0, a1);
  float e0 = __expf(a0 - m), e1 = __expf(a1 - m);
  mode0[i] = e0 / (e0 + e1);
}

__global__ __launch_bounds__(256) void mix_kernel(
    const float* __restrict__ conv, const float* __restrict__ divv,
    const float* __restrict__ mode0,
    float* __restrict__ o32, _Float16* __restrict__ o16)
{
  int idx = blockIdx.x * blockDim.x + threadIdx.x;
  if (idx >= cB * cT * cH) return;
  int bt = idx >> 8;
  float m0 = mode0[bt];
  float v = m0 * conv[idx] + (1.0f - m0) * divv[idx];
  o32[idx] = v; o16[idx] = (_Float16)v;
}

__global__ __launch_bounds__(256) void attn_scores(
    const float* __restrict__ d, const float* __restrict__ e,
    const float* __restrict__ bvec, const float* __restrict__ vvec,
    float* __restrict__ scores)   // (B,T,S); one wave per (b,t,s)
{
  int wid = (blockIdx.x * blockDim.x + threadIdx.x) >> 5;
  if (wid >= cB * cT * cS) return;
  int lane = threadIdx.x & 31;
  int s = wid & 255, bt = wid >> 8, b = bt >> 6;
  const float* drow = d + (long)bt * cH;
  const float* erow = e + (long)(b * cS + s) * cH;
  float acc = 0.0f;
  #pragma unroll
  for (int j = 0; j < 8; ++j) {
    int h = lane + 32 * j;
    acc += tanhf(drow[h] + erow[h] + bvec[h]) * vvec[h];
  }
  #pragma unroll
  for (int m = 16; m >= 1; m >>= 1) acc += __shfl_xor(acc, m, 32);
  if (lane == 0) scores[wid] = acc;
}

__global__ __launch_bounds__(256) void softmax_scale(
    float* __restrict__ data, int N, const float* __restrict__ gate)
{
  long row = blockIdx.x;
  float* p = data + row * N;
  __shared__ float tmp[8];
  int tid = threadIdx.x, lane = tid & 31, w = tid >> 5;
  float mx = -1e30f;
  for (int i = tid; i < N; i += 256) mx = fmaxf(mx, p[i]);
  for (int m = 16; m >= 1; m >>= 1) mx = fmaxf(mx, __shfl_xor(mx, m, 32));
  if (lane == 0) tmp[w] = mx;
  __syncthreads();
  mx = fmaxf(fmaxf(fmaxf(tmp[0], tmp[1]), fmaxf(tmp[2], tmp[3])),
             fmaxf(fmaxf(tmp[4], tmp[5]), fmaxf(tmp[6], tmp[7])));
  __syncthreads();
  float sum = 0.0f;
  for (int i = tid; i < N; i += 256) { float e = __expf(p[i] - mx); p[i] = e; sum += e; }
  for (int m = 16; m >= 1; m >>= 1) sum += __shfl_xor(sum, m, 32);
  if (lane == 0) tmp[w] = sum;
  __syncthreads();
  sum = tmp[0] + tmp[1] + tmp[2] + tmp[3] + tmp[4] + tmp[5] + tmp[6] + tmp[7];
  float sc = (gate ? (1.0f - gate[row]) : 1.0f) / sum;
  for (int i = tid; i < N; i += 256) p[i] *= sc;
}

__global__ __launch_bounds__(256) void encT_kernel(
    const float* __restrict__ enc32, _Float16* __restrict__ encT)
{
  int idx = blockIdx.x * blockDim.x + threadIdx.x;   // (b,h,s)
  if (idx >= cB * cH * cS) return;
  int s = idx & 255, bh = idx >> 8;
  int h = bh & 255, b = bh >> 8;
  encT[idx] = (_Float16)enc32[((long)b * cS + s) * cH + h];
}

__global__ __launch_bounds__(256) void gate_kernel(
    const float* __restrict__ dh2, const float* __restrict__ cctx,
    const float* __restrict__ gW, const float* __restrict__ gb,
    float* __restrict__ gate)
{
  int wid = (blockIdx.x * blockDim.x + threadIdx.x) >> 5;
  if (wid >= cB * cT) return;
  int lane = threadIdx.x & 31;
  float acc = 0.0f;
  #pragma unroll
  for (int j = 0; j < 16; ++j) {
    int h = lane + 32 * j;
    float x = (h < cH) ? dh2[(long)wid * cH + h] : cctx[(long)wid * cH + (h - cH)];
    acc += x * gW[h];
  }
  for (int m = 16; m >= 1; m >>= 1) acc += __shfl_xor(acc, m, 32);
  if (lane == 0) gate[wid] = sigf(acc + gb[0]);
}

__global__ __launch_bounds__(256) void concat_kernel(
    const float* __restrict__ a, const float* __restrict__ b,
    _Float16* __restrict__ out)  // (B*T, 2H)
{
  int idx = blockIdx.x * blockDim.x + threadIdx.x;
  if (idx >= cB * cT * 2 * cH) return;
  int c = idx & 511, bt = idx >> 9;
  float v = (c < cH) ? a[(long)bt * cH + c] : b[(long)bt * cH + (c - cH)];
  out[idx] = (_Float16)v;
}

__global__ __launch_bounds__(256) void scatter_copy(
    const float* __restrict__ cw, const float* __restrict__ gate,
    const int* __restrict__ ids, float* __restrict__ out)
{
  int idx = blockIdx.x * blockDim.x + threadIdx.x;   // (b,t,s)
  if (idx >= cB * cT * cS) return;
  int s = idx & 255, bt = idx >> 8, b = bt >> 6;
  int tok = ids[b * cS + s];
  atomicAdd(&out[(long)bt * cV + tok], gate[bt] * cw[idx]);
}

// ---------------- host side -----------------------------------------------------
static inline int cdiv(int a, int b) { return (a + b - 1) / b; }

static void run_gemm(hipStream_t st, const _Float16* A, long sA, const _Float16* W, long sW,
                     const float* bias, void* C, long sC, int M, int N, int K,
                     int batch, int relu, int half)
{
  int grps = (M / 16) * (N / 64);          // one wave per 16x64 strip
  dim3 grid(cdiv(grps, 8), batch);
  if (relu) {
    if (half) gemm_wmma<1, 1><<<grid, 256, 0, st>>>(A, sA, W, sW, bias, C, sC, M, N, K);
    else      gemm_wmma<1, 0><<<grid, 256, 0, st>>>(A, sA, W, sW, bias, C, sC, M, N, K);
  } else {
    if (half) gemm_wmma<0, 1><<<grid, 256, 0, st>>>(A, sA, W, sW, bias, C, sC, M, N, K);
    else      gemm_wmma<0, 0><<<grid, 256, 0, st>>>(A, sA, W, sW, bias, C, sC, M, N, K);
  }
}

extern "C" void kernel_launch(void* const* d_in, const int* in_sizes, int n_in,
                              void* d_out, int out_size, void* d_ws, size_t ws_size,
                              hipStream_t stream)
{
  (void)in_sizes; (void)n_in; (void)out_size; (void)ws_size;
  const int*   input_ids  = (const int*)  d_in[0];
  const int*   target_ids = (const int*)  d_in[1];
  const float* embedding  = (const float*)d_in[2];
  const float* enc_f_Wih  = (const float*)d_in[3];
  const float* enc_f_Whh  = (const float*)d_in[4];
  const float* enc_f_b    = (const float*)d_in[5];
  const float* enc_b_Wih  = (const float*)d_in[6];
  const float* enc_b_Whh  = (const float*)d_in[7];
  const float* enc_b_b    = (const float*)d_in[8];
  const float* dec_Wih    = (const float*)d_in[9];
  const float* dec_Whh    = (const float*)d_in[10];
  const float* dec_b      = (const float*)d_in[11];
  const float* attn_W     = (const float*)d_in[12];
  const float* attn_b     = (const float*)d_in[13];
  const float* attn_v     = (const float*)d_in[14];
  const float* cattn_W    = (const float*)d_in[15];
  const float* cattn_b    = (const float*)d_in[16];
  const float* cattn_v    = (const float*)d_in[17];
  const float* gate_W     = (const float*)d_in[18];
  const float* gate_b     = (const float*)d_in[19];
  const float* conv1_W    = (const float*)d_in[20];
  const float* conv1_b    = (const float*)d_in[21];
  const float* conv2_W    = (const float*)d_in[22];
  const float* conv2_b    = (const float*)d_in[23];
  const float* div1_W     = (const float*)d_in[24];
  const float* div1_b     = (const float*)d_in[25];
  const float* div2_W     = (const float*)d_in[26];
  const float* div2_b     = (const float*)d_in[27];
  const float* mode_W     = (const float*)d_in[28];
  const float* mode_b     = (const float*)d_in[29];
  const float* out_W      = (const float*)d_in[30];
  const float* out_b      = (const float*)d_in[31];

  char* p = (char*)d_ws;
  auto alloc = [&](size_t bytes) -> void* {
    void* r = (void*)p; p += (bytes + 255) & ~(size_t)255; return r;
  };

  float*    z0     = (float*)   alloc((size_t)cB * cH * 4);
  _Float16* wfih   = (_Float16*)alloc((size_t)cG * cH * 2);
  _Float16* wfhh   = (_Float16*)alloc((size_t)cG * cH * 2);
  _Float16* wbih   = (_Float16*)alloc((size_t)cG * cH * 2);
  _Float16* wbhh   = (_Float16*)alloc((size_t)cG * cH * 2);
  _Float16* wdih   = (_Float16*)alloc((size_t)cG * cH * 2);
  _Float16* wdhh   = (_Float16*)alloc((size_t)cG * cH * 2);
  _Float16* wd1    = (_Float16*)alloc((size_t)cH * cH * 2);
  _Float16* we1    = (_Float16*)alloc((size_t)cH * cH * 2);
  _Float16* wd2    = (_Float16*)alloc((size_t)cH * cH * 2);
  _Float16* we2    = (_Float16*)alloc((size_t)cH * cH * 2);
  _Float16* c1w    = (_Float16*)alloc((size_t)128 * 256 * 2);
  _Float16* c2w    = (_Float16*)alloc((size_t)256 * 128 * 2);
  _Float16* d1w    = (_Float16*)alloc((size_t)512 * 256 * 2);
  _Float16* d2w    = (_Float16*)alloc((size_t)256 * 512 * 2);
  _Float16* outw   = (_Float16*)alloc((size_t)cV * 512 * 2);
  _Float16* xs16   = (_Float16*)alloc((size_t)cS * cB * cH * 2);
  _Float16* tgt16  = (_Float16*)alloc((size_t)cT * cB * cH * 2);
  float*    gxf    = (float*)   alloc((size_t)cS * cB * cG * 4);
  float*    gxb    = (float*)   alloc((size_t)cS * cB * cG * 4);
  float*    gxd    = (float*)   alloc((size_t)cT * cB * cG * 4);
  float*    hf     = (float*)   alloc((size_t)cS * cB * cH * 4);
  float*    hbseq  = (float*)   alloc((size_t)cS * cB * cH * 4);
  float*    dechs  = (float*)   alloc((size_t)cT * cB * cH * 4);
  float*    hbfin  = (float*)   alloc((size_t)cB * cH * 4);
  float*    cbfin  = (float*)   alloc((size_t)cB * cH * 4);
  float*    enc32  = (float*)   alloc((size_t)cB * cS * cH * 4);
  _Float16* enc16  = (_Float16*)alloc((size_t)cB * cS * cH * 2);
  _Float16* encT   = (_Float16*)alloc((size_t)cB * cH * cS * 2);
  float*    dh32   = (float*)   alloc((size_t)cB * cT * cH * 4);
  _Float16* dh16   = (_Float16*)alloc((size_t)cB * cT * cH * 2);
  _Float16* convh  = (_Float16*)alloc((size_t)cB * cT * 128 * 2);
  float*    conv32 = (float*)   alloc((size_t)cB * cT * cH * 4);
  _Float16* divh   = (_Float16*)alloc((size_t)cB * cT * 512 * 2);
  float*    div32  = (float*)   alloc((size_t)cB * cT * cH * 4);
  float*    mode0  = (float*)   alloc((size_t)cB * cT * 4);
  float*    dh2_32 = (float*)   alloc((size_t)cB * cT * cH * 4);
  _Float16* dh2_16 = (_Float16*)alloc((size_t)cB * cT * cH * 2);
  float*    dbuf   = (float*)   alloc((size_t)cB * cT * cH * 4);
  float*    ebuf   = (float*)   alloc((size_t)cB * cS * cH * 4);
  float*    w1     = (float*)   alloc((size_t)cB * cT * cS * 4);
  _Float16* w1h    = (_Float16*)alloc((size_t)cB * cT * cS * 2);
  float*    ctx1   = (float*)   alloc((size_t)cB * cT * cH * 4);
  float*    w2     = (float*)   alloc((size_t)cB * cT * cS * 4);
  _Float16* w2h    = (_Float16*)alloc((size_t)cB * cT * cS * 2);
  float*    cctx   = (float*)   alloc((size_t)cB * cT * cH * 4);
  float*    gateb  = (float*)   alloc((size_t)cB * cT * 4);
  _Float16* comb16 = (_Float16*)alloc((size_t)cB * cT * 2 * cH * 2);

  // ---- prep: zeros + f16 weight conversions + embedding gathers
  fill_zero<<<cdiv(cB * cH, 256), 256, 0, stream>>>(z0, cB * cH);
  cvt_f16<<<cdiv(cG * cH, 256), 256, 0, stream>>>(enc_f_Wih, wfih, cG, cH, cH);
  cvt_f16<<<cdiv(cG * cH, 256), 256, 0, stream>>>(enc_f_Whh, wfhh, cG, cH, cH);
  cvt_f16<<<cdiv(cG * cH, 256), 256, 0, stream>>>(enc_b_Wih, wbih, cG, cH, cH);
  cvt_f16<<<cdiv(cG * cH, 256), 256, 0, stream>>>(enc_b_Whh, wbhh, cG, cH, cH);
  cvt_f16<<<cdiv(cG * cH, 256), 256, 0, stream>>>(dec_Wih, wdih, cG, cH, cH);
  cvt_f16<<<cdiv(cG * cH, 256), 256, 0, stream>>>(dec_Whh, wdhh, cG, cH, cH);
  cvt_f16<<<cdiv(cH * cH, 256), 256, 0, stream>>>(attn_W, wd1, cH, cH, 2 * cH);
  cvt_f16<<<cdiv(cH * cH, 256), 256, 0, stream>>>(attn_W + cH, we1, cH, cH, 2 * cH);
  cvt_f16<<<cdiv(cH * cH, 256), 256, 0, stream>>>(cattn_W, wd2, cH, cH, 2 * cH);
  cvt_f16<<<cdiv(cH * cH, 256), 256, 0, stream>>>(cattn_W + cH, we2, cH, cH, 2 * cH);
  cvt_f16<<<cdiv(128 * 256, 256), 256, 0, stream>>>(conv1_W, c1w, 128, 256, 256);
  cvt_f16<<<cdiv(256 * 128, 256), 256, 0, stream>>>(conv2_W, c2w, 256, 128, 128);
  cvt_f16<<<cdiv(512 * 256, 256), 256, 0, stream>>>(div1_W, d1w, 512, 256, 256);
  cvt_f16<<<cdiv(256 * 512, 256), 256, 0, stream>>>(div2_W, d2w, 256, 512, 512);
  cvt_f16<<<cdiv(cV * 512, 256), 256, 0, stream>>>(out_W, outw, cV, 512, 512);
  embed_gather<<<cdiv(cS * cB * cH, 256), 256, 0, stream>>>(input_ids, embedding, xs16, cS);
  embed_gather<<<cdiv(cT * cB * cH, 256), 256, 0, stream>>>(target_ids, embedding, tgt16, cT);

  // ---- encoder: precompute input gates (parallel WMMA), then persistent recurrences
  run_gemm(stream, xs16, 0, wfih, 0, enc_f_b, gxf, 0, cS * cB, cG, cH, 1, 0, 0);
  run_gemm(stream, xs16, 0, wbih, 0, enc_b_b, gxb, 0, cS * cB, cG, cH, 1, 0, 0);
  lstm_wmma<<<1, 256, 0, stream>>>(gxf, wfhh, z0, z0, hf, nullptr, nullptr, cS, 0);
  lstm_wmma<<<1, 256, 0, stream>>>(gxb, wbhh, z0, z0, hbseq, hbfin, cbfin, cS, 1);
  enc_combine<<<cdiv(cB * cS * cH, 256), 256, 0, stream>>>(hf, hbseq, enc32, enc16);

  // ---- decoder LSTM (init from backward encoder final state)
  run_gemm(stream, tgt16, 0, wdih, 0, dec_b, gxd, 0, cT * cB, cG, cH, 1, 0, 0);
  lstm_wmma<<<1, 256, 0, stream>>>(gxd, wdhh, hbfin, cbfin, dechs, nullptr, nullptr, cT, 0);
  tbh_transpose<<<cdiv(cB * cT * cH, 256), 256, 0, stream>>>(dechs, dh32, dh16);

  // ---- conv / div branches + mode mixing
  run_gemm(stream, dh16, 0, c1w, 0, conv1_b, convh, 0, cB * cT, 128, cH, 1, 1, 1);
  run_gemm(stream, convh, 0, c2w, 0, conv2_b, conv32, 0, cB * cT, cH, 128, 1, 0, 0);
  run_gemm(stream, dh16, 0, d1w, 0, div1_b, divh, 0, cB * cT, 512, cH, 1, 1, 1);
  run_gemm(stream, divh, 0, d2w, 0, div2_b, div32, 0, cB * cT, cH, 512, 1, 0, 0);
  mode_kernel<<<cdiv(cB * cT, 256), 256, 0, stream>>>(dh32, mode_W, mode_b, mode0);
  mix_kernel<<<cdiv(cB * cT * cH, 256), 256, 0, stream>>>(conv32, div32, mode0, dh2_32, dh2_16);

  // ---- shared transposed encoder copy for context GEMMs
  encT_kernel<<<cdiv(cB * cH * cS, 256), 256, 0, stream>>>(enc32, encT);

  // ---- attention 1 (vocab context)
  run_gemm(stream, dh2_16, 0, wd1, 0, nullptr, dbuf, 0, cB * cT, cH, cH, 1, 0, 0);
  run_gemm(stream, enc16, 0, we1, 0, nullptr, ebuf, 0, cB * cS, cH, cH, 1, 0, 0);
  attn_scores<<<cdiv(cB * cT * cS, 8), 256, 0, stream>>>(dbuf, ebuf, attn_b, attn_v, w1);
  softmax_scale<<<cB * cT, 256, 0, stream>>>(w1, cS, nullptr);
  cvt_f16<<<cdiv(cB * cT * cS, 256), 256, 0, stream>>>(w1, w1h, cB * cT, cS, cS);
  run_gemm(stream, w1h, (long)cT * cS, encT, (long)cH * cS, nullptr,
           ctx1, (long)cT * cH, cT, cH, cS, cB, 0, 0);

  // ---- attention 2 (copy context)
  run_gemm(stream, dh2_16, 0, wd2, 0, nullptr, dbuf, 0, cB * cT, cH, cH, 1, 0, 0);
  run_gemm(stream, enc16, 0, we2, 0, nullptr, ebuf, 0, cB * cS, cH, cH, 1, 0, 0);
  attn_scores<<<cdiv(cB * cT * cS, 8), 256, 0, stream>>>(dbuf, ebuf, cattn_b, cattn_v, w2);
  softmax_scale<<<cB * cT, 256, 0, stream>>>(w2, cS, nullptr);
  cvt_f16<<<cdiv(cB * cT * cS, 256), 256, 0, stream>>>(w2, w2h, cB * cT, cS, cS);
  run_gemm(stream, w2h, (long)cT * cS, encT, (long)cH * cS, nullptr,
           cctx, (long)cT * cH, cT, cH, cS, cB, 0, 0);

  // ---- gate, vocab distribution (directly in d_out), copy mechanism
  gate_kernel<<<cdiv(cB * cT, 8), 256, 0, stream>>>(dh2_32, cctx, gate_W, gate_b, gateb);
  concat_kernel<<<cdiv(cB * cT * 2 * cH, 256), 256, 0, stream>>>(dh2_32, ctx1, comb16);
  run_gemm(stream, comb16, 0, outw, 0, out_b, d_out, 0, cB * cT, cV, 2 * cH, 1, 0, 0);
  softmax_scale<<<cB * cT, 256, 0, stream>>>((float*)d_out, cV, gateb);       // *(1-gate)
  scatter_copy<<<cdiv(cB * cT * cS, 256), 256, 0, stream>>>(w2, gateb, input_ids,
                                                            (float*)d_out);   // += gate*cw
}
